// MultiHeadAttention_17308718203673
// MI455X (gfx1250) — compile-verified, matrix operand order checked by reading, not run on hardware
//
#include <hip/hip_runtime.h>
#include <hip/hip_bf16.h>
#include <math.h>

// ---------------------------------------------------------------------------
// Problem constants (match reference)
// ---------------------------------------------------------------------------
constexpr int Bb   = 2;
constexpr int S    = 2048;
constexpr int D    = 1024;
constexpr int H    = 16;
constexpr int DH   = 64;
constexpr int MTOT = Bb * S;          // 4096 rows for the big GEMMs

// ---------------------------------------------------------------------------
// Feature detection for CDNA5 async/tensor data movement
// ---------------------------------------------------------------------------
#if defined(__has_builtin)
#if __has_builtin(__builtin_amdgcn_tensor_load_to_lds)
#define HAVE_TDM 1
#endif
#if __has_builtin(__builtin_amdgcn_global_load_async_to_lds_b128)
#define HAVE_ASYNC 1
#endif
#endif

typedef __attribute__((address_space(3))) void as3_void;
typedef __attribute__((address_space(1))) void as1_void;

// ---------------------------------------------------------------------------
// CDNA5 WMMA types & helpers (wave32, 16x16x32 bf16 -> f32)
// ---------------------------------------------------------------------------
typedef __attribute__((ext_vector_type(16))) __bf16 v16bf;
typedef __attribute__((ext_vector_type(8)))  __bf16 v8bf;
typedef __attribute__((ext_vector_type(8)))  float  v8f;
typedef __attribute__((ext_vector_type(4)))  int    v4i;

__device__ __forceinline__ v8f wmma_bf16(v16bf a, v16bf b, v8f c) {
  // (neg_a, A, neg_b, B, c_mod, C, reuse_a, reuse_b)
  return __builtin_amdgcn_wmma_f32_16x16x32_bf16(false, a, false, b,
                                                 (short)0, c, false, false);
}

// Fragment fetch from an LDS tile stored row-major with K contiguous.
// Per ISA layout: lanes 0-15 hold K=0..7,16..23; lanes 16-31 hold
// K=8..15,24..31 -> two contiguous 16B chunks per lane (2x ds_load_b128).
__device__ __forceinline__ v16bf ldfrag(const __bf16* rowp, int lane) {
  const int hoff = (lane >> 4) * 8;
  v8bf lo = *(const v8bf*)(rowp + hoff);
  v8bf hi = *(const v8bf*)(rowp + 16 + hoff);
  return __builtin_shufflevector(lo, hi, 0, 1, 2, 3, 4, 5, 6, 7,
                                         8, 9, 10, 11, 12, 13, 14, 15);
}

// Convert 16 fp32 -> 16 bf16 and store to LDS (two 16B stores).
__device__ __forceinline__ void cvt_store16(__bf16* dst, float4 f0, float4 f1,
                                            float4 f2, float4 f3) {
  v8bf p0 = {(__bf16)f0.x, (__bf16)f0.y, (__bf16)f0.z, (__bf16)f0.w,
             (__bf16)f1.x, (__bf16)f1.y, (__bf16)f1.z, (__bf16)f1.w};
  v8bf p1 = {(__bf16)f2.x, (__bf16)f2.y, (__bf16)f2.z, (__bf16)f2.w,
             (__bf16)f3.x, (__bf16)f3.y, (__bf16)f3.z, (__bf16)f3.w};
  *(v8bf*)(dst)     = p0;
  *(v8bf*)(dst + 8) = p1;
}

// ---------------------------------------------------------------------------
// Async / TDM helpers
// ---------------------------------------------------------------------------
#ifdef HAVE_ASYNC
// b128 variant takes typed int4 pointers: (addrspace(1) v4i*, addrspace(3) v4i*,
// imm offset, imm cpol) -- per hipcc's own diagnostic of the prototype.
typedef __attribute__((address_space(1))) v4i gv4i;
typedef __attribute__((address_space(3))) v4i lv4i;

__device__ __forceinline__ void async_cp16(const void* g, void* l) {
  __builtin_amdgcn_global_load_async_to_lds_b128((gv4i*)g, (lv4i*)l, 0, 0);
}
__device__ __forceinline__ void wait_async0() {
#if __has_builtin(__builtin_amdgcn_s_wait_asynccnt)
  __builtin_amdgcn_s_wait_asynccnt(0);
#else
  asm volatile("s_wait_asynccnt 0" ::: "memory");
#endif
}
#endif

#ifdef HAVE_TDM
typedef __attribute__((ext_vector_type(4))) unsigned int u32x4;
typedef __attribute__((ext_vector_type(8))) int          i32x8;
typedef __attribute__((ext_vector_type(4))) int          i32x4;

// One 2-D bf16 tile (rows x cols, row stride in elements) -> contiguous LDS.
// D# built per CDNA5 ISA ch.8: group0 = {count/type, lds_addr, gaddr},
// group1 = {data_size, tensor dims, tile dims, dim0 stride}.
__device__ __forceinline__ void tdm_load_2d_bf16(const __bf16* gsrc,
                                                 unsigned lds_off, int rows,
                                                 int cols, int stride_elems) {
  unsigned long long ga = (unsigned long long)(uintptr_t)gsrc;
  u32x4 g0;
  g0[0] = 1u;                                             // count=1, user
  g0[1] = lds_off;                                        // LDS byte addr
  g0[2] = (unsigned)(ga & 0xffffffffu);                   // gaddr[31:0]
  g0[3] = (unsigned)((ga >> 32) & 0x01ffffffu) | (2u << 30);  // gaddr hi, type=2
  i32x8 g1;
  g1[0] = (1 << 16);                                      // data_size=1 -> 2B
  g1[1] = (cols & 0xffff) << 16;                          // tensor_dim0 lo
  g1[2] = ((unsigned)cols >> 16) | ((rows & 0xffff) << 16);  // d0 hi | d1 lo
  g1[3] = ((unsigned)rows >> 16) | ((cols & 0xffff) << 16);  // d1 hi | tile_d0
  g1[4] = (rows & 0xffff);                                // tile_d1 | tile_d2=0
  g1[5] = stride_elems;                                   // dim0 stride lo32
  g1[6] = 0;
  g1[7] = 0;
  i32x4 zz = {0, 0, 0, 0};
#if __clang_major__ >= 23
  i32x8 z8 = {0, 0, 0, 0, 0, 0, 0, 0};
  __builtin_amdgcn_tensor_load_to_lds(g0, g1, zz, zz, z8, 0);
#else
  __builtin_amdgcn_tensor_load_to_lds(g0, g1, zz, zz, 0);
#endif
}
#endif

// ---------------------------------------------------------------------------
// Kernel 1: P[m,n] = sum_k X[m,k]*W[n,k] + bias[n]   (fp32 in, bf16 out)
// M=4096, N=1024, K=1024. Tile 128x128x32, 256 thr = 8 waves (2x4 frags each)
// ---------------------------------------------------------------------------
__global__ __launch_bounds__(256) void proj_kernel(
    const float* __restrict__ X, const float* __restrict__ W,
    const float* __restrict__ bias, __bf16* __restrict__ P) {
  __shared__ __bf16 As[128 * 32];
  __shared__ __bf16 Bs[128 * 32];

  const int t = threadIdx.x, lane = t & 31, wave = t >> 5;
  const int wm = wave >> 1, wn = wave & 1;           // 4x2 wave grid
  const int m0 = blockIdx.x * 128, n0 = blockIdx.y * 128;
  const int row = t >> 1, seg = (t & 1) * 16;        // staging: 2 thr / row

  v8f acc[2][4];
#pragma unroll
  for (int i = 0; i < 2; i++)
#pragma unroll
    for (int j = 0; j < 4; j++) acc[i][j] = {};

  for (int kb = 0; kb < D; kb += 32) {
    const float4* xs = (const float4*)(X + (size_t)(m0 + row) * D + kb + seg);
    cvt_store16(&As[row * 32 + seg], xs[0], xs[1], xs[2], xs[3]);
    const float4* ws = (const float4*)(W + (size_t)(n0 + row) * D + kb + seg);
    cvt_store16(&Bs[row * 32 + seg], ws[0], ws[1], ws[2], ws[3]);
    __syncthreads();

    if (kb + 32 < D) {  // global_prefetch_b8 for next K tile
      __builtin_prefetch(X + (size_t)(m0 + row) * D + kb + 32 + seg, 0, 0);
      __builtin_prefetch(W + (size_t)(n0 + row) * D + kb + 32 + seg, 0, 0);
    }

    v16bf af[2], bfr[4];
#pragma unroll
    for (int fm = 0; fm < 2; fm++)
      af[fm] = ldfrag(&As[(wm * 32 + fm * 16 + (lane & 15)) * 32], lane);
#pragma unroll
    for (int fn = 0; fn < 4; fn++)
      bfr[fn] = ldfrag(&Bs[(wn * 64 + fn * 16 + (lane & 15)) * 32], lane);
#pragma unroll
    for (int fm = 0; fm < 2; fm++)
#pragma unroll
      for (int fn = 0; fn < 4; fn++)
        acc[fm][fn] = wmma_bf16(af[fm], bfr[fn], acc[fm][fn]);
    __syncthreads();
  }

  const int nlo = lane & 15, hh = lane >> 4;
#pragma unroll
  for (int fm = 0; fm < 2; fm++)
#pragma unroll
    for (int fn = 0; fn < 4; fn++) {
      const int n = n0 + wn * 64 + fn * 16 + nlo;
      const float bv = bias[n];
#pragma unroll
      for (int r = 0; r < 8; r++) {
        const int m = m0 + wm * 32 + fm * 16 + hh * 8 + r;
        P[(size_t)m * D + n] = (__bf16)(acc[fm][fn][r] + bv);
      }
    }
}

// ---------------------------------------------------------------------------
// Kernel 2: scores[z,m,n] = (qh[m,:] . kh[n,:]) / 8   per z=b*H+h
// M=N=2048, K=64. bf16 in. Q/K tiles staged with the Tensor Data Mover.
// ---------------------------------------------------------------------------
__global__ __launch_bounds__(256) void scores_kernel(
    const __bf16* __restrict__ QP, const __bf16* __restrict__ KP,
    float* __restrict__ ATTN) {
  __shared__ __bf16 Qs[128 * 64];
  __shared__ __bf16 Ks[128 * 64];

  const int t = threadIdx.x, lane = t & 31, wave = t >> 5;
  const int wm = wave >> 1, wn = wave & 1;
  const int z = blockIdx.z, bi = z >> 4, hi = z & 15;
  const int m0 = blockIdx.x * 128, n0 = blockIdx.y * 128;
  const __bf16* qbase = QP + (size_t)bi * S * D + hi * DH;
  const __bf16* kbase = KP + (size_t)bi * S * D + hi * DH;

#ifdef HAVE_TDM
  if (wave == 0) {
    unsigned q_lds = (unsigned)(uintptr_t)(as3_void*)&Qs[0];
    unsigned k_lds = (unsigned)(uintptr_t)(as3_void*)&Ks[0];
    tdm_load_2d_bf16(qbase + (size_t)m0 * D, q_lds, 128, 64, D);
    tdm_load_2d_bf16(kbase + (size_t)n0 * D, k_lds, 128, 64, D);
    __builtin_amdgcn_s_wait_tensorcnt(0);
    asm volatile("" ::: "memory");  // TDM wrote LDS behind the compiler's back
  }
#else
  {
    const int row = t >> 1, seg = (t & 1) * 32;  // 2 thr/row, 32 bf16 each
    const uint4* qs = (const uint4*)(qbase + (size_t)(m0 + row) * D + seg);
    uint4* qd = (uint4*)(&Qs[row * 64 + seg]);
    qd[0] = qs[0]; qd[1] = qs[1]; qd[2] = qs[2]; qd[3] = qs[3];
    const uint4* ks = (const uint4*)(kbase + (size_t)(n0 + row) * D + seg);
    uint4* kd = (uint4*)(&Ks[row * 64 + seg]);
    kd[0] = ks[0]; kd[1] = ks[1]; kd[2] = ks[2]; kd[3] = ks[3];
  }
#endif
  __syncthreads();

  v8f acc[2][4];
#pragma unroll
  for (int i = 0; i < 2; i++)
#pragma unroll
    for (int j = 0; j < 4; j++) acc[i][j] = {};

#pragma unroll
  for (int kt = 0; kt < DH; kt += 32) {
    v16bf af[2], bfr[4];
#pragma unroll
    for (int fm = 0; fm < 2; fm++)
      af[fm] = ldfrag(&Qs[(wm * 32 + fm * 16 + (lane & 15)) * 64 + kt], lane);
#pragma unroll
    for (int fn = 0; fn < 4; fn++)
      bfr[fn] = ldfrag(&Ks[(wn * 64 + fn * 16 + (lane & 15)) * 64 + kt], lane);
#pragma unroll
    for (int fm = 0; fm < 2; fm++)
#pragma unroll
      for (int fn = 0; fn < 4; fn++)
        acc[fm][fn] = wmma_bf16(af[fm], bfr[fn], acc[fm][fn]);
  }

  const int nlo = lane & 15, hh = lane >> 4;
  float* arow = ATTN + (size_t)z * S * S;
#pragma unroll
  for (int fm = 0; fm < 2; fm++)
#pragma unroll
    for (int fn = 0; fn < 4; fn++) {
      const int n = n0 + wn * 64 + fn * 16 + nlo;
#pragma unroll
      for (int r = 0; r < 8; r++) {
        const int m = m0 + wm * 32 + fm * 16 + hh * 8 + r;
        arow[(size_t)m * S + n] = acc[fm][fn][r] * 0.125f;  // 1/sqrt(64)
      }
    }
}

// ---------------------------------------------------------------------------
// Kernel 3: in-place row softmax over 2048-wide rows of ATTN.
// ---------------------------------------------------------------------------
__global__ __launch_bounds__(256) void softmax_kernel(float* __restrict__ ATTN) {
  __shared__ float red[256];
  float* rowp = ATTN + (size_t)blockIdx.x * S;
  const int t = threadIdx.x;

  float x[8];
  float mx = -INFINITY;
#pragma unroll
  for (int i = 0; i < 8; i++) {
    x[i] = rowp[t + i * 256];
    mx = fmaxf(mx, x[i]);
  }
  red[t] = mx;
  __syncthreads();
  for (int off = 128; off > 0; off >>= 1) {
    if (t < off) red[t] = fmaxf(red[t], red[t + off]);
    __syncthreads();
  }
  mx = red[0];
  __syncthreads();

  float s = 0.f;
#pragma unroll
  for (int i = 0; i < 8; i++) {
    x[i] = __expf(x[i] - mx);
    s += x[i];
  }
  red[t] = s;
  __syncthreads();
  for (int off = 128; off > 0; off >>= 1) {
    if (t < off) red[t] += red[t + off];
    __syncthreads();
  }
  const float inv = 1.0f / red[0];
#pragma unroll
  for (int i = 0; i < 8; i++) rowp[t + i * 256] = x[i] * inv;
}

// ---------------------------------------------------------------------------
// Kernel 4: ctx[m,d] = sum_k attn[z,m,k]*vh[k,d], bf16 out in merged-head
// layout CC[(b*S+m)*D + h*64 + d]. M=2048, N=64, K=2048. Tile 128x64x32.
// ---------------------------------------------------------------------------
__global__ __launch_bounds__(256) void ctx_kernel(
    const float* __restrict__ ATTN, const __bf16* __restrict__ VP,
    __bf16* __restrict__ CC) {
  __shared__ __bf16 As[128 * 32];
  __shared__ __bf16 Vs[64 * 32];  // transposed: Vs[n*32 + k]

  const int t = threadIdx.x, lane = t & 31, wave = t >> 5;
  const int wm = wave >> 1, wn = wave & 1;  // 4x2
  const int z = blockIdx.z, bi = z >> 4, hi = z & 15;
  const int m0 = blockIdx.x * 128;
  const float*  abase = ATTN + (size_t)z * S * S;
  const __bf16* vbase = VP + (size_t)bi * S * D + hi * DH;
  const int row = t >> 1, seg = (t & 1) * 16;
  const int vk = t >> 3, vn = (t & 7) * 8;  // V staging: 8 bf16/thread

  v8f acc[2][2];
#pragma unroll
  for (int i = 0; i < 2; i++)
#pragma unroll
    for (int j = 0; j < 2; j++) acc[i][j] = {};

  for (int kb = 0; kb < S; kb += 32) {
    const float4* asrc = (const float4*)(abase + (size_t)(m0 + row) * S + kb + seg);
    cvt_store16(&As[row * 32 + seg], asrc[0], asrc[1], asrc[2], asrc[3]);
    {  // stage V with transpose (k-major -> n-major so frag K is contiguous)
      v8bf vv = *(const v8bf*)(vbase + (size_t)(kb + vk) * D + vn);
#pragma unroll
      for (int j = 0; j < 8; j++) Vs[(vn + j) * 32 + vk] = vv[j];
    }
    __syncthreads();

    if (kb + 32 < S)
      __builtin_prefetch(abase + (size_t)(m0 + row) * S + kb + 32 + seg, 0, 0);

    v16bf af[2], bfr[2];
#pragma unroll
    for (int fm = 0; fm < 2; fm++)
      af[fm] = ldfrag(&As[(wm * 32 + fm * 16 + (lane & 15)) * 32], lane);
#pragma unroll
    for (int fn = 0; fn < 2; fn++)
      bfr[fn] = ldfrag(&Vs[(wn * 32 + fn * 16 + (lane & 15)) * 32], lane);
#pragma unroll
    for (int fm = 0; fm < 2; fm++)
#pragma unroll
      for (int fn = 0; fn < 2; fn++)
        acc[fm][fn] = wmma_bf16(af[fm], bfr[fn], acc[fm][fn]);
    __syncthreads();
  }

  const int nlo = lane & 15, hh = lane >> 4;
#pragma unroll
  for (int fm = 0; fm < 2; fm++)
#pragma unroll
    for (int fn = 0; fn < 2; fn++) {
      const int n = wn * 32 + fn * 16 + nlo;  // 0..63 within head
#pragma unroll
      for (int r = 0; r < 8; r++) {
        const int m = m0 + wm * 32 + fm * 16 + hh * 8 + r;
        CC[((size_t)bi * S + m) * D + hi * DH + n] = (__bf16)acc[fm][fn][r];
      }
    }
}

// ---------------------------------------------------------------------------
// Kernel 5: out[m,n] = sum_k CC[m,k]*Wo[n,k] + bo[n]. bf16 A staged with
// GLOBAL_LOAD_ASYNC_TO_LDS_B128 (ASYNCcnt), fp32 W converted in staging.
// ---------------------------------------------------------------------------
__global__ __launch_bounds__(256) void out_kernel(
    const __bf16* __restrict__ CC, const float* __restrict__ W,
    const float* __restrict__ bias, float* __restrict__ OUT) {
  __shared__ __bf16 As[128 * 32];
  __shared__ __bf16 Bs[128 * 32];

  const int t = threadIdx.x, lane = t & 31, wave = t >> 5;
  const int wm = wave >> 1, wn = wave & 1;
  const int m0 = blockIdx.x * 128, n0 = blockIdx.y * 128;
  const int row = t >> 1, seg = (t & 1) * 16;

  v8f acc[2][4];
#pragma unroll
  for (int i = 0; i < 2; i++)
#pragma unroll
    for (int j = 0; j < 4; j++) acc[i][j] = {};

  for (int kb = 0; kb < D; kb += 32) {
#ifdef HAVE_ASYNC
    async_cp16(CC + (size_t)(m0 + row) * D + kb + seg, &As[row * 32 + seg]);
    async_cp16(CC + (size_t)(m0 + row) * D + kb + seg + 8,
               &As[row * 32 + seg + 8]);
#else
    {
      const uint4* as = (const uint4*)(CC + (size_t)(m0 + row) * D + kb + seg);
      uint4* ad = (uint4*)(&As[row * 32 + seg]);
      ad[0] = as[0];
      ad[1] = as[1];
    }
#endif
    const float4* ws = (const float4*)(W + (size_t)(n0 + row) * D + kb + seg);
    cvt_store16(&Bs[row * 32 + seg], ws[0], ws[1], ws[2], ws[3]);
#ifdef HAVE_ASYNC
    wait_async0();
#endif
    __syncthreads();

    if (kb + 32 < D) {
      __builtin_prefetch(CC + (size_t)(m0 + row) * D + kb + 32 + seg, 0, 0);
      __builtin_prefetch(W + (size_t)(n0 + row) * D + kb + 32 + seg, 0, 0);
    }

    v16bf af[2], bfr[4];
#pragma unroll
    for (int fm = 0; fm < 2; fm++)
      af[fm] = ldfrag(&As[(wm * 32 + fm * 16 + (lane & 15)) * 32], lane);
#pragma unroll
    for (int fn = 0; fn < 4; fn++)
      bfr[fn] = ldfrag(&Bs[(wn * 64 + fn * 16 + (lane & 15)) * 32], lane);
#pragma unroll
    for (int fm = 0; fm < 2; fm++)
#pragma unroll
      for (int fn = 0; fn < 4; fn++)
        acc[fm][fn] = wmma_bf16(af[fm], bfr[fn], acc[fm][fn]);
    __syncthreads();
  }

  const int nlo = lane & 15, hh = lane >> 4;
#pragma unroll
  for (int fm = 0; fm < 2; fm++)
#pragma unroll
    for (int fn = 0; fn < 4; fn++) {
      const int n = n0 + wn * 64 + fn * 16 + nlo;
      const float bv = bias[n];
#pragma unroll
      for (int r = 0; r < 8; r++) {
        const int m = m0 + wm * 32 + fm * 16 + hh * 8 + r;
        OUT[(size_t)m * D + n] = acc[fm][fn][r] + bv;
      }
    }
}

// ---------------------------------------------------------------------------
// Host launcher
// ---------------------------------------------------------------------------
extern "C" void kernel_launch(void* const* d_in, const int* in_sizes, int n_in,
                              void* d_out, int out_size, void* d_ws,
                              size_t ws_size, hipStream_t stream) {
  (void)in_sizes; (void)n_in; (void)out_size; (void)ws_size;

  const float* q   = (const float*)d_in[0];
  const float* k   = (const float*)d_in[1];
  const float* v   = (const float*)d_in[2];
  const float* w_q = (const float*)d_in[3];
  const float* b_q = (const float*)d_in[4];
  const float* w_k = (const float*)d_in[5];
  const float* b_k = (const float*)d_in[6];
  const float* w_v = (const float*)d_in[7];
  const float* b_v = (const float*)d_in[8];
  const float* w_o = (const float*)d_in[9];
  const float* b_o = (const float*)d_in[10];

  float* out  = (float*)d_out;                       // [B,S,D]
  float* attn = out + (size_t)MTOT * D;              // [B,H,S,S]

  const size_t PLANE = (size_t)MTOT * D;             // 4M elements
  __bf16* qp = (__bf16*)d_ws;
  __bf16* kp = qp + PLANE;
  __bf16* vp = kp + PLANE;
  __bf16* cc = vp + PLANE;                            // merged-head ctx

  dim3 blk(256);

  dim3 gproj(MTOT / 128, D / 128);                   // 32 x 8
  proj_kernel<<<gproj, blk, 0, stream>>>(q, w_q, b_q, qp);
  proj_kernel<<<gproj, blk, 0, stream>>>(k, w_k, b_k, kp);
  proj_kernel<<<gproj, blk, 0, stream>>>(v, w_v, b_v, vp);

  dim3 gsc(S / 128, S / 128, Bb * H);                // 16 x 16 x 32
  scores_kernel<<<gsc, blk, 0, stream>>>(qp, kp, attn);

  softmax_kernel<<<dim3(Bb * H * S), blk, 0, stream>>>(attn);

  dim3 gctx(S / 128, 1, Bb * H);                     // 16 x 1 x 32
  ctx_kernel<<<gctx, blk, 0, stream>>>(attn, vp, cc);

  dim3 gout(MTOT / 128, D / 128);                    // 32 x 8
  out_kernel<<<gout, blk, 0, stream>>>(cc, w_o, b_o, out);
}
